// Attention_64338610094364
// MI455X (gfx1250) — matrix operand order repaired and checked by reading, not compiled, not run on hardware
//
#include <hip/hip_runtime.h>
#include <hip/hip_bf16.h>
#include <stdint.h>

#define B_ 64
#define S_ 2048
#define D_ 1024   // decoder dim (N of main GEMM)
#define M_ 1024   // memory dim  (K of main GEMM)

// ---- main kernel tiling ----
#define MTILE 128                 // s-rows per workgroup
#define NTILE 128                 // d-cols per n-chunk
#define KBLK  64                  // k rows staged per B block
#define ASTRIDE 2064              // bytes per A row in LDS: 1024 bf16 + 16B pad (stride%256==16 -> conflict free)
#define BSTRIDE 272               // bytes per B row in LDS: 128  bf16 + 16B pad
#define ABYTES (MTILE * ASTRIDE)  // 264192
#define BBYTES (KBLK * BSTRIDE)   // 17408
#define NSTAGE (D_ / NTILE * M_ / KBLK)      // 128 staging phases
#define SMEM_BYTES (ABYTES + 2 * BBYTES)     // 299008 <= 320KB/WG (CDNA5)

typedef __bf16 v8bf  __attribute__((ext_vector_type(8)));
typedef __bf16 v16bf __attribute__((ext_vector_type(16)));
typedef float  v8f   __attribute__((ext_vector_type(8)));

__device__ __forceinline__ unsigned short f2bf_rne(float f) {
  unsigned u = __float_as_uint(f);
  unsigned r = u + 0x7fffu + ((u >> 16) & 1u);   // round-to-nearest-even
  return (unsigned short)(r >> 16);
}

__device__ __forceinline__ float fast_tanh(float x) {
  x = fminf(fmaxf(x, -15.f), 15.f);              // avoid inf*0 -> NaN
  float e = __expf(2.f * x);                     // v_exp_f32
  return (e - 1.f) * __builtin_amdgcn_rcpf(e + 1.f);
}

// ---------------- kernel 1: dec_feat = decoder_state @ W_dec  [64,1024]
__global__ void dec_proj_kernel(const float* __restrict__ dec,
                                const float* __restrict__ Wdec,
                                float* __restrict__ decf) {
  int b = blockIdx.y;
  int d = blockIdx.x * 256 + threadIdx.x;
  const float* dr = dec + b * D_;
  float acc = 0.f;
  #pragma unroll 4
  for (int m = 0; m < D_; ++m) acc = fmaf(dr[m], Wdec[m * D_ + d], acc);
  decf[b * D_ + d] = acc;
}

// ---------------- kernel 2: W_mem f32 -> bf16 (row-major, K-major for WMMA B)
__global__ void wconv_kernel(const float* __restrict__ W,
                             unsigned short* __restrict__ Wb) {
  int i = blockIdx.x * 256 + threadIdx.x;        // float4 index
  float4 f = ((const float4*)W)[i];
  ushort4 o;
  o.x = f2bf_rne(f.x); o.y = f2bf_rne(f.y);
  o.z = f2bf_rne(f.z); o.w = f2bf_rne(f.w);
  ((ushort4*)Wb)[i] = o;
}

// ---------------- kernel 3: fused  scores = (tanh(mem@Wmem + dec_feat)) . W_v
__global__ __launch_bounds__(256, 1)
void attn_scores_kernel(const float* __restrict__ mem,          // [B,S,M] f32
                        const unsigned short* __restrict__ Wb,  // [M,D] bf16
                        const float* __restrict__ decf,         // [B,D]
                        const float* __restrict__ wv,           // [D]
                        float* __restrict__ scores) {           // [B,S]
  extern __shared__ char smem[];
  char* Abuf = smem;                    // 128 x 1024 bf16, padded rows
  // B ping/pong buffers live at smem + ABYTES (+ BBYTES)

  const int b      = blockIdx.y;
  const int s0     = blockIdx.x * MTILE;
  const int tid    = threadIdx.x;
  const int lane   = tid & 31;
  const int w      = tid >> 5;
  const int mw     = w & 3;       // 4 m-waves, 32 rows each
  const int nw     = w >> 2;      // 2 n-waves, 64 cols each
  const int laneLo = lane & 15;
  const int hi     = lane >> 4;

  // DS byte offset of dynamic LDS base (low 32 bits of shared-aperture addr)
  const unsigned bB0  = (unsigned)(uintptr_t)smem + ABYTES;
  const uint64_t wb64 = (uint64_t)(uintptr_t)Wb;

  // Issue one B-block stage (64 K-rows x 128 N-cols, bf16) as async DMA to LDS.
  // Each of the 256 threads moves 4 x 16B;  ASYNCcnt-tracked, in-order.
  auto issue_stage = [&](int t) {
    const int nc = t >> 4;
    const int kb = t & 15;
    #pragma unroll
    for (int it = 0; it < 4; ++it) {
      int fl = tid + it * 256;          // 0..1023
      int kr = fl >> 4;                 // 0..63
      int c  = fl & 15;                 // 16B chunk within 256B row
      unsigned lds  = bB0 + (unsigned)(t & 1) * BBYTES + kr * BSTRIDE + c * 16;
      unsigned goff = (unsigned)(((kb * KBLK + kr) * D_ + nc * NTILE) * 2 + c * 16);
      asm volatile("global_load_async_to_lds_b128 %0, %1, %2"
                   :: "v"(lds), "v"(goff), "s"(wb64) : "memory");
    }
  };

  issue_stage(0);   // overlap first B stage with the A-tile conversion below

  // ---- stage A tile once: 128x1024 f32 -> bf16 LDS (coalesced float4 loads)
  {
    const float* src = mem + ((size_t)(b * S_ + s0)) * (size_t)M_;
    #pragma unroll 4
    for (int it = 0; it < (MTILE * M_ / 4) / 256; ++it) {   // 128 iters
      int f4  = tid + it * 256;
      int row = f4 >> 8;                                     // 256 float4/row
      int c4  = f4 & 255;
      float4 v = ((const float4*)src)[row * 256 + c4];
      union { unsigned short us[4]; uint2 u2; } p;
      p.us[0] = f2bf_rne(v.x); p.us[1] = f2bf_rne(v.y);
      p.us[2] = f2bf_rne(v.z); p.us[3] = f2bf_rne(v.w);
      *(uint2*)(Abuf + row * ASTRIDE + c4 * 8) = p.u2;
    }
  }

  float scoreAcc[2][8];
  #pragma unroll
  for (int i = 0; i < 2; ++i)
    #pragma unroll
    for (int v = 0; v < 8; ++v) scoreAcc[i][v] = 0.f;

  const float* decb = decf + b * D_;

  for (int nc = 0; nc < D_ / NTILE; ++nc) {                  // 8 n-chunks
    v8f acc[2][4];
    #pragma unroll
    for (int i = 0; i < 2; ++i)
      #pragma unroll
      for (int j = 0; j < 4; ++j) { v8f z = {}; acc[i][j] = z; }

    for (int kb = 0; kb < M_ / KBLK; ++kb) {                 // 16 k-blocks
      const int t = nc * (M_ / KBLK) + kb;
      // buffer[(t+1)&1] is free once everyone finished compute(t-1)
      __syncthreads();
      int tn = t + 1; if (tn > NSTAGE - 1) tn = NSTAGE - 1;  // clamped re-issue
      issue_stage(tn);                                       // prefetch next block
      // async loads complete in order: <=4 outstanding => stage t has landed
      asm volatile("s_wait_asynccnt 0x4" ::: "memory");
      __syncthreads();                                       // all waves' stage t done
      const char* Bcur = smem + ABYTES + (t & 1) * BBYTES;

      #pragma unroll
      for (int ks = 0; ks < 2; ++ks) {                       // k-steps of 32
        const int k0 = kb * KBLK + ks * 32;
        // A fragments: lane<16 holds K {0..7,16..23}, lane>=16 holds {8..15,24..31}
        v16bf a[2];
        #pragma unroll
        for (int i = 0; i < 2; ++i) {
          int row = mw * 32 + i * 16 + laneLo;
          const char* pa = Abuf + row * ASTRIDE + (k0 + hi * 8) * 2;
          v8bf c0 = *(const v8bf*)pa;
          v8bf c1 = *(const v8bf*)(pa + 32);
          a[i] = __builtin_shufflevector(c0, c1,
                   0,1,2,3,4,5,6,7,8,9,10,11,12,13,14,15);
        }
        // B fragments: lane holds column N=laneLo, K {0..7,16..23} (lane<16) / {8..15,24..31} (lane>=16)
        #pragma unroll
        for (int j = 0; j < 4; ++j) {
          const char* pb = Bcur + (ks * 32 + hi * 8) * BSTRIDE
                                + (nw * 64 + j * 16 + laneLo) * 2;
          v16bf bf;
          #pragma unroll
          for (int s = 0; s < 8; ++s) {
            bf[s]     = *(const __bf16*)(pb + s * BSTRIDE);
            bf[s + 8] = *(const __bf16*)(pb + (s + 16) * BSTRIDE);
          }
          #pragma unroll
          for (int i = 0; i < 2; ++i)
            acc[i][j] = __builtin_amdgcn_wmma_f32_16x16x32_bf16(
                false, a[i], false, bf, (short)0, acc[i][j], false, false);
        }
      }
    }

    // ---- fused epilogue: scoreAcc += tanh(acc + dec_feat[n]) * W_v[n]
    // D layout: element (v,lane) = (M = v + 8*hi, N = laneLo)
    #pragma unroll
    for (int j = 0; j < 4; ++j) {
      int n   = nc * NTILE + nw * 64 + j * 16 + laneLo;
      float dv = decb[n];
      float wn = wv[n];
      #pragma unroll
      for (int i = 0; i < 2; ++i)
        #pragma unroll
        for (int v = 0; v < 8; ++v)
          scoreAcc[i][v] += fast_tanh(acc[i][j][v] + dv) * wn;
    }
  }

  // ---- reduce each row over its 16 lanes, write scores
  #pragma unroll
  for (int i = 0; i < 2; ++i)
    #pragma unroll
    for (int v = 0; v < 8; ++v) {
      float x = scoreAcc[i][v];
      x += __shfl_xor(x, 1, 32);
      x += __shfl_xor(x, 2, 32);
      x += __shfl_xor(x, 4, 32);
      x += __shfl_xor(x, 8, 32);
      if (laneLo == 0) {
        int srow = s0 + mw * 32 + i * 16 + v + 8 * hi;
        scores[b * S_ + srow] = x;
      }
    }
}

// ---------------- kernel 4: masked softmax over S=2048 per batch
__global__ void softmax_kernel(const float* __restrict__ scores,
                               const int* __restrict__ mask,
                               float* __restrict__ attn) {
  __shared__ float red[256];
  int b = blockIdx.x, tid = threadIdx.x;
  float loc[8];
  float mx = -__builtin_inff();
  #pragma unroll
  for (int i = 0; i < 8; ++i) {
    int idx = tid + i * 256;
    float v = scores[b * S_ + idx];
    if (mask[b * S_ + idx] == 0) v = -__builtin_inff();
    loc[i] = v;
    mx = fmaxf(mx, v);
  }
  red[tid] = mx; __syncthreads();
  for (int off = 128; off > 0; off >>= 1) {
    if (tid < off) red[tid] = fmaxf(red[tid], red[tid + off]);
    __syncthreads();
  }
  mx = red[0]; __syncthreads();
  float s = 0.f;
  #pragma unroll
  for (int i = 0; i < 8; ++i) {
    float e = (loc[i] == -__builtin_inff()) ? 0.f : __expf(loc[i] - mx);
    loc[i] = e; s += e;
  }
  red[tid] = s; __syncthreads();
  for (int off = 128; off > 0; off >>= 1) {
    if (tid < off) red[tid] += red[tid + off];
    __syncthreads();
  }
  float inv = 1.0f / red[0];
  #pragma unroll
  for (int i = 0; i < 8; ++i)
    attn[b * S_ + tid + i * 256] = loc[i] * inv;
}

// ---------------- kernel 5: context = attn @ memory_bank  [64,1024]
__global__ void context_kernel(const float* __restrict__ attn,
                               const float* __restrict__ mem,
                               float* __restrict__ ctx) {
  int b = blockIdx.y;
  int m = blockIdx.x * 256 + threadIdx.x;
  const float* a   = attn + b * S_;
  const float* col = mem + (size_t)b * S_ * M_ + m;
  float acc = 0.f;
  #pragma unroll 4
  for (int s = 0; s < S_; ++s)
    acc = fmaf(a[s], col[(size_t)s * M_], acc);
  ctx[b * M_ + m] = acc;
}

extern "C" void kernel_launch(void* const* d_in, const int* in_sizes, int n_in,
                              void* d_out, int out_size, void* d_ws, size_t ws_size,
                              hipStream_t stream) {
  (void)in_sizes; (void)n_in; (void)out_size; (void)ws_size;
  const float* dec  = (const float*)d_in[0];   // [64,1024]
  const float* mem  = (const float*)d_in[1];   // [64,2048,1024]
  const int*   mask = (const int*)d_in[2];     // [64,2048]
  const float* Wv   = (const float*)d_in[3];   // [1024]
  const float* Wdec = (const float*)d_in[4];   // [1024,1024]
  const float* Wmem = (const float*)d_in[5];   // [1024,1024]

  float* out     = (float*)d_out;
  float* context = out;                        // [64,1024]
  float* attn    = out + B_ * D_;              // [64,2048]

  char* ws = (char*)d_ws;
  unsigned short* Wb = (unsigned short*)ws;                          // 2 MB
  float* decf   = (float*)(ws + (size_t)M_ * D_ * 2);                // 256 KB
  float* scores = (float*)(ws + (size_t)M_ * D_ * 2
                              + (size_t)B_ * D_ * 4);                // 512 KB

  hipFuncSetAttribute(reinterpret_cast<const void*>(attn_scores_kernel),
                      hipFuncAttributeMaxDynamicSharedMemorySize, SMEM_BYTES);

  dec_proj_kernel<<<dim3(D_ / 256, B_), 256, 0, stream>>>(dec, Wdec, decf);
  wconv_kernel<<<dim3((M_ * D_ / 4) / 256), 256, 0, stream>>>(Wmem, Wb);
  attn_scores_kernel<<<dim3(S_ / MTILE, B_), 256, SMEM_BYTES, stream>>>(
      mem, Wb, decf, Wv, scores);
  softmax_kernel<<<dim3(B_), 256, 0, stream>>>(scores, mask, attn);
  context_kernel<<<dim3(M_ / 256, B_), 256, 0, stream>>>(attn, mem, context);
}
